// SimPool_58342835748995
// MI455X (gfx1250) — matrix operand order repaired and checked by reading, not compiled, not run on hardware
//
#include <hip/hip_runtime.h>
#include <hip/hip_bf16.h>
#include <math.h>

// Problem constants (from reference setup)
static constexpr int kB = 16;
static constexpr int kN = 8192;
static constexpr int kD = 768;
static constexpr float kScale = 0.03608439182435161f;   // 1/sqrt(768)
static constexpr float kEpsLN = 1e-6f;
static constexpr float kEpsPool = 1e-6f;

typedef __attribute__((ext_vector_type(16))) __bf16 v16bf;
typedef __attribute__((ext_vector_type(8)))  float  v8f;

// ---- order-preserving float<->uint encoding for global atomic min ----
__device__ __forceinline__ unsigned encf(float f) {
    unsigned u = __float_as_uint(f);
    return (u & 0x80000000u) ? ~u : (u | 0x80000000u);
}
__device__ __forceinline__ float decf(unsigned e) {
    return __uint_as_float((e & 0x80000000u) ? (e ^ 0x80000000u) : ~e);
}

// =====================================================================
// Kernel 1: q[b,d] = mean_n x[b,n,d]   (partial sums via atomicAdd)
// grid (32 n-chunks, 3 d-chunks, 16 b), block 256
// =====================================================================
__global__ __launch_bounds__(256)
void colmean_kernel(const float* __restrict__ x, float* __restrict__ q) {
    const int b  = blockIdx.z;
    const int di = blockIdx.y * 256 + threadIdx.x;
    const int n0 = blockIdx.x * 256;
    const float* xp = x + ((long long)b * kN + n0) * kD + di;
    float s = 0.f;
    #pragma unroll 4
    for (int n = 0; n < 256; ++n) s += xp[(long long)n * kD];
    atomicAdd(q + b * kD + di, s * (1.0f / (float)kN));
}

// =====================================================================
// Kernel 2: D[16 x 768] = A[16 x 768] * B   via v_wmma_f32_16x16x32_bf16
//   BTRANS=1: B[k][n] = W[(nt+n)*768 + k]   (for qq = q @ wq^T)
//   BTRANS=0: B[k][n] = W[k*768 + nt + n]   (for r  = qq @ wk)
// BTRANS is a compile-time template parameter so both address paths are
// straight-line code (no per-element branches in the WMMA inner loop).
// One wave per 16-wide output tile; grid 48 blocks x 32 threads.
// A-matrix lane layout (16-bit, 16x32): lane m=L&15, half hi=L>>4,
//   element e -> K = e + 16*hi (this code's order, not the ISA table order)
// B-matrix: N = L&15, K = e + 16*hi (same order as A).   C/D: M = vgpr + 8*hi, N = L&15.
// =====================================================================
template <int BTRANS>
__global__ __launch_bounds__(32)
void proj16_wmma_kernel(const float* __restrict__ A, const float* __restrict__ W,
                        float* __restrict__ Dout) {
    const int lane = threadIdx.x;
    const int nt   = blockIdx.x * 16;
    const int ln   = lane & 15;       // A row m / B,C,D column n
    const int hi   = lane >> 4;

    v8f acc = {};
    for (int k0 = 0; k0 < kD; k0 += 32) {
        v16bf a, bm;
        #pragma unroll
        for (int e = 0; e < 16; ++e) {
            const int K = e + hi * 16;
            a[e] = (__bf16)A[ln * kD + k0 + K];
        }
        const int koff = k0 + hi * 16;
        #pragma unroll
        for (int e = 0; e < 16; ++e) {
            const int k = koff + e;
            const float w = BTRANS ? W[(nt + ln) * kD + k]   // contiguous in e
                                   : W[k * kD + nt + ln];    // coalesced across lanes
            bm[e] = (__bf16)w;
        }
        // D = A x B + C   (8 args: neg_a, A, neg_b, B, c_mod, C, reuse_a, reuse_b)
        acc = __builtin_amdgcn_wmma_f32_16x16x32_bf16(
                  false, a, false, bm, (short)0, acc, false, false);
    }
    #pragma unroll
    for (int v = 0; v < 8; ++v) {
        const int mrow = v + hi * 8;
        Dout[mrow * kD + nt + ln] = acc[v];
    }
}

// =====================================================================
// Kernel 3: per-row LN stats + logits + global kv-min.
// One wave (32 lanes) per row of 768; block = 8 waves; grid = B*N/8.
// =====================================================================
__global__ __launch_bounds__(256)
void rowpass_kernel(const float* __restrict__ x, const float* __restrict__ lnw,
                    const float* __restrict__ lnb, const float* __restrict__ r,
                    float* __restrict__ mu_o, float* __restrict__ rstd_o,
                    float* __restrict__ logits, unsigned* __restrict__ gmin) {
    const int lane = threadIdx.x & 31;
    const int wave = threadIdx.x >> 5;
    const long long row  = (long long)blockIdx.x * 8 + wave;   // < B*N
    const int       b    = (int)(row / kN);
    const long long base = row * kD;

    float vals[kD / 32];
    float s = 0.f, ss = 0.f;
    #pragma unroll
    for (int i = 0; i < kD / 32; ++i) {
        const float v = x[base + lane + i * 32];
        vals[i] = v;
        s  += v;
        ss += v * v;
    }
    #pragma unroll
    for (int off = 16; off; off >>= 1) {
        s  += __shfl_xor(s,  off, 32);
        ss += __shfl_xor(ss, off, 32);
    }
    const float mu   = s * (1.0f / (float)kD);
    const float var  = ss * (1.0f / (float)kD) - mu * mu;
    const float rstd = rsqrtf(var + kEpsLN);

    const float* rb = r + b * kD;
    float dot = 0.f, mn = 3.402823466e38f;
    #pragma unroll
    for (int i = 0; i < kD / 32; ++i) {
        const int   j  = lane + i * 32;
        const float kv = (vals[i] - mu) * rstd * lnw[j] + lnb[j];
        dot = fmaf(kv, rb[j], dot);
        mn  = fminf(mn, kv);
    }
    #pragma unroll
    for (int off = 16; off; off >>= 1) {
        dot += __shfl_xor(dot, off, 32);
        mn   = fminf(mn, __shfl_xor(mn, off, 32));
    }

    __shared__ float smn[8];
    if (lane == 0) {
        mu_o[row]   = mu;
        rstd_o[row] = rstd;
        logits[row] = dot * kScale;
        smn[wave]   = mn;
    }
    __syncthreads();
    if (threadIdx.x == 0) {
        float m2 = smn[0];
        #pragma unroll
        for (int i = 1; i < 8; ++i) m2 = fminf(m2, smn[i]);
        atomicMin(gmin, encf(m2));
    }
}

// =====================================================================
// Kernel 4: per-batch softmax stats (row max + sum of exp). grid = B.
// =====================================================================
__global__ __launch_bounds__(256)
void softmax_stats_kernel(const float* __restrict__ logits,
                          float* __restrict__ mout, float* __restrict__ zout) {
    const int b = blockIdx.x;
    const float* l = logits + (long long)b * kN;
    __shared__ float sm[256];

    float m = -3.402823466e38f;
    for (int i = threadIdx.x; i < kN; i += 256) m = fmaxf(m, l[i]);
    sm[threadIdx.x] = m;
    __syncthreads();
    for (int s = 128; s; s >>= 1) {
        if (threadIdx.x < s) sm[threadIdx.x] = fmaxf(sm[threadIdx.x], sm[threadIdx.x + s]);
        __syncthreads();
    }
    m = sm[0];
    __syncthreads();

    float z = 0.f;
    for (int i = threadIdx.x; i < kN; i += 256) z += __expf(l[i] - m);
    sm[threadIdx.x] = z;
    __syncthreads();
    for (int s = 128; s; s >>= 1) {
        if (threadIdx.x < s) sm[threadIdx.x] += sm[threadIdx.x + s];
        __syncthreads();
    }
    if (threadIdx.x == 0) { mout[b] = m; zout[b] = sm[0]; }
}

// =====================================================================
// Kernel 5: out_acc[b,d] += sum_n attn[b,n] * (kv[b,n,d] - gmin + eps)^2
// grid (32 n-chunks, 3 d-chunks, 16 b), block 256.
// =====================================================================
__global__ __launch_bounds__(256)
void accum_kernel(const float* __restrict__ x, const float* __restrict__ lnw,
                  const float* __restrict__ lnb, const float* __restrict__ mu,
                  const float* __restrict__ rstd, const float* __restrict__ logits,
                  const float* __restrict__ mv, const float* __restrict__ zv,
                  const unsigned* __restrict__ gmin, float* __restrict__ out_acc) {
    const int b  = blockIdx.z;
    const int di = blockIdx.y * 256 + threadIdx.x;
    const int n0 = blockIdx.x * 256;

    __shared__ float wn[256], mus[256], rsd[256];
    {
        const long long row = (long long)b * kN + n0 + threadIdx.x;
        wn[threadIdx.x]  = __expf(logits[row] - mv[b]) / zv[b];
        mus[threadIdx.x] = mu[row];
        rsd[threadIdx.x] = rstd[row];
    }
    __syncthreads();

    const float g  = decf(*gmin);
    const float w_ = lnw[di];
    const float b_ = lnb[di];
    const float* xp = x + ((long long)b * kN + n0) * kD + di;

    float acc = 0.f;
    #pragma unroll 4
    for (int n = 0; n < 256; ++n) {
        if (n + 8 < 256) __builtin_prefetch(xp + (long long)(n + 8) * kD, 0, 0);
        const float kv  = (xp[(long long)n * kD] - mus[n]) * rsd[n] * w_ + b_;
        const float val = kv - g + kEpsPool;
        acc = fmaf(wn[n] * val, val, acc);
    }
    atomicAdd(out_acc + b * kD + di, acc);
}

// =====================================================================
// Kernel 6: out = sqrt(out_acc)    (GeM pooling root, gamma = 2)
// =====================================================================
__global__ __launch_bounds__(256)
void finalize_kernel(const float* __restrict__ acc, float* __restrict__ out) {
    const int i = blockIdx.x * 256 + threadIdx.x;
    if (i < kB * kD) out[i] = sqrtf(acc[i]);
}

// =====================================================================
// Launch
// =====================================================================
extern "C" void kernel_launch(void* const* d_in, const int* in_sizes, int n_in,
                              void* d_out, int out_size, void* d_ws, size_t ws_size,
                              hipStream_t stream) {
    const float* x   = (const float*)d_in[0];   // [16, 8192, 768]
    const float* lnw = (const float*)d_in[1];   // [768]
    const float* lnb = (const float*)d_in[2];   // [768]
    const float* wq  = (const float*)d_in[3];   // [768, 768]
    const float* wk  = (const float*)d_in[4];   // [768, 768]
    float* out = (float*)d_out;                 // [16, 768]

    // workspace layout (floats); total ~1.77 MB
    float* ws = (float*)d_ws;
    float*    q       = ws;                         // 12288
    float*    qq      = q       + kB * kD;          // 12288
    float*    r       = qq      + kB * kD;          // 12288
    float*    mu      = r       + kB * kD;          // 131072
    float*    rstd    = mu      + kB * kN;          // 131072
    float*    logits  = rstd    + kB * kN;          // 131072
    unsigned* gmin    = (unsigned*)(logits + kB * kN); // 1 (+pad)
    float*    mv      = (float*)gmin + 16;          // 16
    float*    zv      = mv      + kB;               // 16
    float*    out_acc = zv      + kB;               // 12288

    hipMemsetAsync(q,       0,    kB * kD * sizeof(float), stream);
    hipMemsetAsync(out_acc, 0,    kB * kD * sizeof(float), stream);
    hipMemsetAsync(gmin,    0xFF, sizeof(unsigned),        stream);

    // 1) q = mean_n x
    colmean_kernel<<<dim3(kN / 256, kD / 256, kB), 256, 0, stream>>>(x, q);
    // 2) qq = q @ wq^T ; r = qq @ wk      (WMMA bf16, M=16 tiles)
    proj16_wmma_kernel<1><<<kD / 16, 32, 0, stream>>>(q,  wq, qq);
    proj16_wmma_kernel<0><<<kD / 16, 32, 0, stream>>>(qq, wk, r);
    // 3) per-row LN stats + logits + global kv min
    rowpass_kernel<<<(kB * kN) / 8, 256, 0, stream>>>(x, lnw, lnb, r,
                                                      mu, rstd, logits, gmin);
    // 4) softmax stats per batch
    softmax_stats_kernel<<<kB, 256, 0, stream>>>(logits, mv, zv);
    // 5) attention-weighted GeM accumulation
    accum_kernel<<<dim3(kN / 256, kD / 256, kB), 256, 0, stream>>>(
        x, lnw, lnb, mu, rstd, logits, mv, zv, gmin, out_acc);
    // 6) sqrt
    finalize_kernel<<<(kB * kD + 255) / 256, 256, 0, stream>>>(out_acc, out);
}
